// graph_gru_sage_7851200217451
// MI455X (gfx1250) — matrix operand order repaired and checked by reading, not compiled, not run on hardware
//
#include <hip/hip_runtime.h>

// ---------------- types ----------------
typedef __bf16 bf16_t;
typedef __attribute__((ext_vector_type(16))) __bf16 bf16x16;
typedef __attribute__((ext_vector_type(8)))  __bf16 bf16x8;
typedef __attribute__((ext_vector_type(8)))  float  f32x8;

#define HDIM 64

// ---------------- helpers ----------------
__device__ __forceinline__ unsigned short f2bf(float f) {
    union { float f; unsigned u; } v; v.f = f;
    unsigned u = v.u;
    u += 0x7FFFu + ((u >> 16) & 1u);            // round-to-nearest-even
    return (unsigned short)(u >> 16);
}

__device__ __forceinline__ f32x8 wmma_bf16(bf16x16 a, bf16x16 b, f32x8 c) {
    // D = A(16x32 bf16) * B(32x16 bf16) + C(16x16 f32)
    return __builtin_amdgcn_wmma_f32_16x16x32_bf16(false, a, false, b, (short)0, c, false, false);
}

// A fragment: rows r0..r0+15, K-chunk [koff, koff+32) of a row-major [*, 64] bf16 matrix.
// ISA layout: lanes 0-15 -> M=lane, K=koff+0..7 (elems 0..7) and koff+16..23 (elems 8..15);
//             lanes 16-31 -> M=lane-16, K=koff+8..15 and koff+24..31.
__device__ __forceinline__ bf16x16 load_a_frag(const bf16_t* A, int r0, int koff, int lane) {
    int m  = lane & 15;
    int hi = lane >> 4;
    const bf16_t* p = A + (size_t)(r0 + m) * HDIM + koff + hi * 8;
    bf16x8 lo = *(const bf16x8*)(p);
    bf16x8 up = *(const bf16x8*)(p + 16);
    bf16x16 a;
#pragma unroll
    for (int i = 0; i < 8; ++i) { a[i] = lo[i]; a[i + 8] = up[i]; }
    return a;
}

// B fragment for Y = X @ W^T : B[k][n] = W[n][k], W row-major 64x64 bf16.
// Lane reads W row (c0 + lane%16), k = koff + hi*8 + 0..7 and koff + 16 + hi*8 + 0..7 (same order as A).
__device__ __forceinline__ bf16x16 load_b_frag(const bf16_t* W, int c0, int koff, int lane) {
    int n  = lane & 15;
    int hi = lane >> 4;
    const bf16_t* p = W + (size_t)(c0 + n) * HDIM + koff + hi * 8;
    bf16x8 lo = *(const bf16x8*)(p);
    bf16x8 up = *(const bf16x8*)(p + 16);
    bf16x16 b;
#pragma unroll
    for (int i = 0; i < 8; ++i) { b[i] = lo[i]; b[i + 8] = up[i]; }
    return b;
}

// ---------------- small kernels ----------------
__global__ void fill_zero_f32(float* p, int n) {
    int i = blockIdx.x * blockDim.x + threadIdx.x;
    if (i < n) p[i] = 0.0f;
}

__global__ void deg_kernel(const long long* dst, float* deg, int E) {
    int e = blockIdx.x * blockDim.x + threadIdx.x;
    if (e < E) atomicAdd(&deg[(int)dst[e]], 1.0f);
}

__global__ void invdeg_kernel(float* deg, int N) {
    int n = blockIdx.x * blockDim.x + threadIdx.x;
    if (n < N) { float d = deg[n]; deg[n] = (d > 0.0f) ? 1.0f / d : 0.0f; }
}

__global__ void cvt_w_kernel(const float* Wl, const float* Wr,
                             unsigned short* Wlb, unsigned short* Wrb, int n) {
    int i = blockIdx.x * blockDim.x + threadIdx.x;
    if (i < n) { Wlb[i] = f2bf(Wl[i]); Wrb[i] = f2bf(Wr[i]); }
}

// one thread per (edge, feature): coalesced gather + f32 atomic scatter, two tensors at once
__global__ void scatter2_kernel(const float* __restrict__ x, const float* __restrict__ h,
                                const long long* __restrict__ src, const long long* __restrict__ dst,
                                float* aggx, float* aggh, int E) {
    int t = blockIdx.x * blockDim.x + threadIdx.x;
    int e = t >> 6, f = t & 63;
    if (e >= E) return;
    int s = (int)src[e], d = (int)dst[e];
    atomicAdd(&aggx[(size_t)d * HDIM + f], x[(size_t)s * HDIM + f]);
    atomicAdd(&aggh[(size_t)d * HDIM + f], h[(size_t)s * HDIM + f]);
}

__global__ void scatter1_kernel(const float* __restrict__ x,
                                const long long* __restrict__ src, const long long* __restrict__ dst,
                                float* agg, int E) {
    int t = blockIdx.x * blockDim.x + threadIdx.x;
    int e = t >> 6, f = t & 63;
    if (e >= E) return;
    int s = (int)src[e], d = (int)dst[e];
    atomicAdd(&agg[(size_t)d * HDIM + f], x[(size_t)s * HDIM + f]);
}

__global__ void cvt_nodes_kernel(const float* aggx, const float* aggh,
                                 const float* x, const float* h, const float* invdeg,
                                 unsigned short* aggxb, unsigned short* agghb,
                                 unsigned short* xb, unsigned short* hb, int total) {
    int i = blockIdx.x * blockDim.x + threadIdx.x;
    if (i >= total) return;
    float id = invdeg[i >> 6];
    aggxb[i] = f2bf(aggx[i] * id);
    agghb[i] = f2bf(aggh[i] * id);
    xb[i]    = f2bf(x[i]);
    hb[i]    = f2bf(h[i]);
}

__global__ void cvt_agg_kernel(const float* agg, const float* invdeg,
                               unsigned short* aggb, int total) {
    int i = blockIdx.x * blockDim.x + threadIdx.x;
    if (i >= total) return;
    aggb[i] = f2bf(agg[i] * invdeg[i >> 6]);
}

// ---------------- fused WMMA kernels ----------------
// Phase B: gates 0..4 for a 16x16 tile per wave; 20 chained v_wmma_f32_16x16x32_bf16.
// Epilogue computes z, r, r*h and gate-4 preactivation in registers.
__global__ __launch_bounds__(256)
void gemm_gates01234_kernel(const unsigned short* aggx_, const unsigned short* x_,
                            const unsigned short* aggh_, const unsigned short* h_,
                            const float* __restrict__ h_f, const float* __restrict__ bias,
                            const unsigned short* Wl_, const unsigned short* Wr_,
                            float* z_out, float* t4_out, float* rh_out,
                            unsigned short* rh_b, int N) {
    const bf16_t* aggx = (const bf16_t*)aggx_;
    const bf16_t* xbf  = (const bf16_t*)x_;
    const bf16_t* aggh = (const bf16_t*)aggh_;
    const bf16_t* hbf  = (const bf16_t*)h_;
    const bf16_t* Wl   = (const bf16_t*)Wl_;
    const bf16_t* Wr   = (const bf16_t*)Wr_;

    int lane = threadIdx.x & 31;
    int wave = threadIdx.x >> 5;                 // 8 waves
    int c0 = (wave & 3) << 4;                    // 4 col-tiles cover all 64 outputs
    int r0 = blockIdx.x * 32 + ((wave >> 2) << 4);
    if (r0 + 16 > N) return;                     // wave-uniform

    bf16x16 ax0 = load_a_frag(aggx, r0, 0, lane), ax1 = load_a_frag(aggx, r0, 32, lane);
    bf16x16 xx0 = load_a_frag(xbf,  r0, 0, lane), xx1 = load_a_frag(xbf,  r0, 32, lane);
    bf16x16 ah0 = load_a_frag(aggh, r0, 0, lane), ah1 = load_a_frag(aggh, r0, 32, lane);
    bf16x16 hh0 = load_a_frag(hbf,  r0, 0, lane), hh1 = load_a_frag(hbf,  r0, 32, lane);

    f32x8 acc[5];
#pragma unroll
    for (int g = 0; g < 5; ++g) {
        const bf16_t* Wlg = Wl + g * (HDIM * HDIM);
        const bf16_t* Wrg = Wr + g * (HDIM * HDIM);
        bool useh = (g == 1) || (g == 3);        // gates on h vs gates on x
        bf16x16 a0  = useh ? ah0 : ax0;
        bf16x16 a1  = useh ? ah1 : ax1;
        bf16x16 ra0 = useh ? hh0 : xx0;
        bf16x16 ra1 = useh ? hh1 : xx1;
        f32x8 c = {};
        c = wmma_bf16(a0,  load_b_frag(Wlg, c0, 0,  lane), c);
        c = wmma_bf16(a1,  load_b_frag(Wlg, c0, 32, lane), c);
        c = wmma_bf16(ra0, load_b_frag(Wrg, c0, 0,  lane), c);
        c = wmma_bf16(ra1, load_b_frag(Wrg, c0, 32, lane), c);
        acc[g] = c;
    }

    int col   = c0 + (lane & 15);
    int rbase = r0 + ((lane >> 4) << 3);
    float b0 = bias[0 * HDIM + col], b1 = bias[1 * HDIM + col];
    float b2 = bias[2 * HDIM + col], b3 = bias[3 * HDIM + col];
    float b4 = bias[4 * HDIM + col];
#pragma unroll
    for (int i = 0; i < 8; ++i) {
        size_t idx = (size_t)(rbase + i) * HDIM + col;
        float g0 = acc[0][i] + b0, g1 = acc[1][i] + b1;
        float g2 = acc[2][i] + b2, g3 = acc[3][i] + b3;
        float g4 = acc[4][i] + b4;
        float zz = 1.0f / (1.0f + __expf(-(g0 + g1)));
        float rr = 1.0f / (1.0f + __expf(-(g2 + g3)));
        float hv = h_f[idx];
        float rh = rr * hv;
        z_out[idx]  = zz;
        t4_out[idx] = g4;
        rh_out[idx] = rh;
        rh_b[idx]   = f2bf(rh);
    }
}

// Phase D: gate 5 + GRU blend; 4 chained WMMAs, writes h_new straight to output.
__global__ __launch_bounds__(256)
void gemm_gate5_kernel(const unsigned short* aggrh_, const unsigned short* rh_,
                       const float* __restrict__ h_f, const float* __restrict__ z_buf,
                       const float* __restrict__ t4_buf, const float* __restrict__ bias5,
                       const unsigned short* Wl5_, const unsigned short* Wr5_,
                       float* out, int N) {
    const bf16_t* aggrh = (const bf16_t*)aggrh_;
    const bf16_t* rhbf  = (const bf16_t*)rh_;
    const bf16_t* Wl5   = (const bf16_t*)Wl5_;
    const bf16_t* Wr5   = (const bf16_t*)Wr5_;

    int lane = threadIdx.x & 31;
    int wave = threadIdx.x >> 5;
    int c0 = (wave & 3) << 4;
    int r0 = blockIdx.x * 32 + ((wave >> 2) << 4);
    if (r0 + 16 > N) return;

    bf16x16 aa0 = load_a_frag(aggrh, r0, 0, lane), aa1 = load_a_frag(aggrh, r0, 32, lane);
    bf16x16 rr0 = load_a_frag(rhbf,  r0, 0, lane), rr1 = load_a_frag(rhbf,  r0, 32, lane);

    f32x8 c = {};
    c = wmma_bf16(aa0, load_b_frag(Wl5, c0, 0,  lane), c);
    c = wmma_bf16(aa1, load_b_frag(Wl5, c0, 32, lane), c);
    c = wmma_bf16(rr0, load_b_frag(Wr5, c0, 0,  lane), c);
    c = wmma_bf16(rr1, load_b_frag(Wr5, c0, 32, lane), c);

    int col   = c0 + (lane & 15);
    int rbase = r0 + ((lane >> 4) << 3);
    float b5 = bias5[col];
#pragma unroll
    for (int i = 0; i < 8; ++i) {
        size_t idx = (size_t)(rbase + i) * HDIM + col;
        float g5 = c[i] + b5;
        float ht = tanhf(t4_buf[idx] + g5);
        float zz = z_buf[idx];
        float hv = h_f[idx];
        out[idx] = zz * hv + (1.0f - zz) * ht;
    }
}

// ---------------- launcher ----------------
extern "C" void kernel_launch(void* const* d_in, const int* in_sizes, int n_in,
                              void* d_out, int out_size, void* d_ws, size_t ws_size,
                              hipStream_t stream) {
    const float*     inp  = (const float*)d_in[0];
    const long long* edg  = (const long long*)d_in[1];   // int64 per reference
    const float*     h_in = (const float*)d_in[2];
    const float*     Wl   = (const float*)d_in[3];
    const float*     b    = (const float*)d_in[4];
    const float*     Wr   = (const float*)d_in[5];
    float* out = (float*)d_out;

    const int N = in_sizes[0] / HDIM;
    const int E = in_sizes[1] / 2;
    const int L = in_sizes[2] / in_sizes[0];
    const size_t NH = (size_t)N * HDIM;
    const int NHi = (int)NH;

    const long long* src = edg;
    const long long* dst = edg + E;

    // workspace carve (256B aligned regions)
    char* w = (char*)d_ws;
    size_t off = 0;
    auto carve = [&](size_t bytes) -> char* {
        char* p = w + off;
        off += (bytes + 255) & ~(size_t)255;
        return p;
    };
    float* invdeg = (float*)carve((size_t)N * 4);
    float* aggx_f = (float*)carve(NH * 4);   // reused for agg(r*h)
    float* aggh_f = (float*)carve(NH * 4);
    float* rh_f   = (float*)carve(NH * 4);
    float* z_f    = (float*)carve(NH * 4);
    float* t4_f   = (float*)carve(NH * 4);
    unsigned short* aggx_b = (unsigned short*)carve(NH * 2);  // reused for agg(r*h) bf16
    unsigned short* aggh_b = (unsigned short*)carve(NH * 2);
    unsigned short* x_b    = (unsigned short*)carve(NH * 2);
    unsigned short* h_b    = (unsigned short*)carve(NH * 2);
    unsigned short* rh_b   = (unsigned short*)carve(NH * 2);
    const int welem = L * 6 * HDIM * HDIM;
    unsigned short* Wl_b = (unsigned short*)carve((size_t)welem * 2);
    unsigned short* Wr_b = (unsigned short*)carve((size_t)welem * 2);
    (void)ws_size; (void)n_in; (void)out_size;

    const int T = 256;
    const unsigned nBlkN  = (unsigned)((N + T - 1) / T);
    const unsigned nBlkE  = (unsigned)((E + T - 1) / T);
    const unsigned nBlkNH = (unsigned)((NHi + T - 1) / T);
    const unsigned nBlkEF = (unsigned)(((long long)E * HDIM + T - 1) / T);
    const unsigned nBlkW  = (unsigned)((welem + T - 1) / T);
    dim3 gemm_grid((unsigned)((N + 31) / 32));

    // shared normalizer + weight conversion (once per call)
    fill_zero_f32<<<nBlkN, T, 0, stream>>>(invdeg, N);
    deg_kernel<<<nBlkE, T, 0, stream>>>(dst, invdeg, E);
    invdeg_kernel<<<nBlkN, T, 0, stream>>>(invdeg, N);
    cvt_w_kernel<<<nBlkW, T, 0, stream>>>(Wl, Wr, Wl_b, Wr_b, welem);

    for (int layer = 0; layer < L; ++layer) {
        const float* x_f  = (layer == 0) ? inp : (out + (size_t)(layer - 1) * NH);
        const float* h_f  = h_in + (size_t)layer * NH;
        const float* bias = b + (size_t)layer * 6 * HDIM;
        const unsigned short* Wl_l = Wl_b + (size_t)layer * 6 * HDIM * HDIM;
        const unsigned short* Wr_l = Wr_b + (size_t)layer * 6 * HDIM * HDIM;
        float* out_l = out + (size_t)layer * NH;

        // Phase A: mean-aggregate x and h
        fill_zero_f32<<<nBlkNH, T, 0, stream>>>(aggx_f, NHi);
        fill_zero_f32<<<nBlkNH, T, 0, stream>>>(aggh_f, NHi);
        scatter2_kernel<<<nBlkEF, T, 0, stream>>>(x_f, h_f, src, dst, aggx_f, aggh_f, E);
        cvt_nodes_kernel<<<nBlkNH, T, 0, stream>>>(aggx_f, aggh_f, x_f, h_f, invdeg,
                                                   aggx_b, aggh_b, x_b, h_b, NHi);
        // Phase B: fused 5-gate WMMA + z/r/rh epilogue
        gemm_gates01234_kernel<<<gemm_grid, T, 0, stream>>>(aggx_b, x_b, aggh_b, h_b, h_f, bias,
                                                            Wl_l, Wr_l, z_f, t4_f, rh_f, rh_b, N);
        // Phase C: aggregate r*h (reuse aggx buffers)
        fill_zero_f32<<<nBlkNH, T, 0, stream>>>(aggx_f, NHi);
        scatter1_kernel<<<nBlkEF, T, 0, stream>>>(rh_f, src, dst, aggx_f, E);
        cvt_agg_kernel<<<nBlkNH, T, 0, stream>>>(aggx_f, invdeg, aggx_b, NHi);
        // Phase D: gate-5 WMMA + tanh + GRU blend -> h_new
        gemm_gate5_kernel<<<gemm_grid, T, 0, stream>>>(aggx_b, rh_b, h_f, z_f, t4_f,
                                                       bias + 5 * HDIM,
                                                       Wl_l + 5 * HDIM * HDIM,
                                                       Wr_l + 5 * HDIM * HDIM,
                                                       out_l, N);
    }
}
